// LOWERNN_81346680586286
// MI455X (gfx1250) — matrix operand order repaired and checked by reading, not compiled, not run on hardware
//
#include <hip/hip_runtime.h>
#include <hip/hip_bf16.h>

// ---------------------------------------------------------------------------
// Problem constants (from the reference)
// ---------------------------------------------------------------------------
#define NSEQ   208          // total ragged sequences
#define LMAX   128          // padded sequence length
#define DMODEL 512          // model dim
#define HDIM   256          // LSTM hidden (per direction)
#define G4H    1024         // 4*H gate dim
#define NHEAD  4
#define DHEAD  128
#define BB     8
#define EE     32
#define MROWS  (NSEQ * LMAX)   // 26624 rows for the big GEMMs

typedef __attribute__((ext_vector_type(16))) __bf16 bf16x16;
typedef __attribute__((ext_vector_type(8)))  float  f32x8;

// ---------------------------------------------------------------------------
// WMMA helpers (CDNA5 V_WMMA_F32_16X16X32_BF16 fragment layouts, ISA §7.12.2)
//   A (16x32, 16-bit): lane r=lane&15 row, h=lane>>4:
//     elems 0..7  = K in [8h,   8h+8)
//     elems 8..15 = K in [16+8h,16+8h+8)
//   B (32x16, 16-bit): lane c=lane&15 col, h=lane>>4:
//     elems 0..7  = K in [8h,   8h+8)
//     elems 8..15 = K in [16+8h,16+8h+8)
//   C/D (f32): VGPR v -> M = v + 8*(lane>=16), N = lane&15
// ---------------------------------------------------------------------------
__device__ __forceinline__ f32x8 wmma_bf16(bf16x16 a, bf16x16 b, f32x8 c) {
  return __builtin_amdgcn_wmma_f32_16x16x32_bf16(false, a, false, b,
                                                 (short)0, c, false, false);
}

__device__ __forceinline__ bf16x16 load_a_frag_bf16(const __bf16* row, int h) {
  bf16x16 a;
  const __bf16* p = row + 8 * h;
#pragma unroll
  for (int i = 0; i < 8; ++i) a[i] = p[i];
  p = row + 16 + 8 * h;
#pragma unroll
  for (int i = 0; i < 8; ++i) a[8 + i] = p[i];
  return a;
}

__device__ __forceinline__ bf16x16 load_a_frag_f32(const float* row, int h) {
  bf16x16 a;
#pragma unroll
  for (int i = 0; i < 8; ++i) a[i] = (__bf16)row[8 * h + i];
#pragma unroll
  for (int i = 0; i < 8; ++i) a[8 + i] = (__bf16)row[16 + 8 * h + i];
  return a;
}

__device__ __forceinline__ bf16x16 load_b_frag_bf16(const __bf16* col, int h) {
  bf16x16 b;
  const __bf16* p = col + 8 * h;
#pragma unroll
  for (int i = 0; i < 8; ++i) b[i] = p[i];
  p = col + 16 + 8 * h;
#pragma unroll
  for (int i = 0; i < 8; ++i) b[8 + i] = p[i];
  return b;
}

__device__ __forceinline__ float sigmf(float x) {
  return 1.0f / (1.0f + __expf(-x));
}

// ---------------------------------------------------------------------------
// Generic WMMA GEMM:  C[M,N] = epi( A[M,K] @ B ),  all dims multiples of
// 128/128/32 for our call sites.
//   btrans=1 : Bsrc is N x K row-major (weight (out,in)),  B[k][n]=Bsrc[n][k]
//   btrans=0 : Bsrc is K x N row-major,                    B[k][n]=Bsrc[k][n]
// Epilogue: v = acc + bias[n];  if(rowmask && !rowmask[row]) v = 0;
//           if(resid) v += resid[row*ldc+n];   C[row*ldc+n] = v;
// Block: 256 threads (8 waves as 2x4), tile 128x128, k-step 32, bf16 LDS.
// ---------------------------------------------------------------------------
__global__ __launch_bounds__(256)
void gemm_wmma(const float* __restrict__ A, int lda,
               const float* __restrict__ Bm, int ldb, int btrans,
               const float* __restrict__ bias,
               const float* __restrict__ resid,
               const int*   __restrict__ rowmask,
               float* __restrict__ C, int ldc,
               int M, int Ncols, int K) {
  __shared__ __bf16 Alds[128][34];
  __shared__ __bf16 Blds[128][34];   // stored as Blds[n][k]
  const int tid  = threadIdx.x;
  const int lane = tid & 31, wave = tid >> 5;
  const int hi = lane >> 4, cn = lane & 15;
  const int wm = wave & 1, wn = wave >> 1;
  const int m0 = blockIdx.y * 128, n0 = blockIdx.x * 128;

  f32x8 acc[4][2];
#pragma unroll
  for (int ms = 0; ms < 4; ++ms)
#pragma unroll
    for (int ns = 0; ns < 2; ++ns)
#pragma unroll
      for (int v = 0; v < 8; ++v) acc[ms][ns][v] = 0.0f;

  for (int k0 = 0; k0 < K; k0 += 32) {
    // stage A tile (128x32) -> bf16
    {
      const int r = tid >> 1, cb = (tid & 1) * 16;
      const float* src = A + (size_t)(m0 + r) * lda + k0 + cb;
      if (k0 + 32 < K) __builtin_prefetch(src + 32, 0, 1);
#pragma unroll
      for (int u = 0; u < 16; ++u) Alds[r][cb + u] = (__bf16)src[u];
    }
    // stage B tile (32x128) -> Blds[n][k]
    if (btrans) {
      const int r = tid >> 1, cb = (tid & 1) * 16;
      const float* src = Bm + (size_t)(n0 + r) * ldb + k0 + cb;
#pragma unroll
      for (int u = 0; u < 16; ++u) Blds[r][cb + u] = (__bf16)src[u];
    } else {
#pragma unroll
      for (int u = 0; u < 16; ++u) {
        int i = tid + u * 256;
        int kk = i >> 7, nn = i & 127;
        Blds[nn][kk] = (__bf16)Bm[(size_t)(k0 + kk) * ldb + n0 + nn];
      }
    }
    __syncthreads();

    bf16x16 af[4];
#pragma unroll
    for (int ms = 0; ms < 4; ++ms)
      af[ms] = load_a_frag_bf16(&Alds[wm * 64 + ms * 16 + cn][0], hi);
#pragma unroll
    for (int ns = 0; ns < 2; ++ns) {
      bf16x16 bf = load_b_frag_bf16(&Blds[wn * 32 + ns * 16 + cn][0], hi);
#pragma unroll
      for (int ms = 0; ms < 4; ++ms)
        acc[ms][ns] = wmma_bf16(af[ms], bf, acc[ms][ns]);
    }
    __syncthreads();
  }

#pragma unroll
  for (int ms = 0; ms < 4; ++ms)
#pragma unroll
    for (int ns = 0; ns < 2; ++ns) {
      const int col = n0 + wn * 32 + ns * 16 + cn;
      const float bv = bias ? bias[col] : 0.0f;
#pragma unroll
      for (int v = 0; v < 8; ++v) {
        const int row = m0 + wm * 64 + ms * 16 + v + 8 * hi;
        float val = acc[ms][ns][v] + bv;
        if (rowmask && rowmask[row] == 0) val = 0.0f;
        if (resid) val += resid[(size_t)row * ldc + col];
        C[(size_t)row * ldc + col] = val;
      }
    }
}

// ---------------------------------------------------------------------------
// Batched weight-stationary LSTM recurrence, 16 sequences per block.
// Per step: G(16x1024) = gather(xg) + H(16x256) @ Whh^T, fully via WMMA.
// Wave w owns column tiles jt in {w, w+8}; their 4 gate tiles (i,f,g,o at
// n = g*256 + jt*16) all live in this wave -> cell update is elementwise on
// the C-fragments in registers (no gate LDS round-trip).
// backward=1 gathers xg at t = len-1-s and scatters h back to t (this exactly
// reproduces reverse_within -> LSTM -> reverse_within).
// ---------------------------------------------------------------------------
__global__ __launch_bounds__(256)
void lstm_rec(const float* __restrict__ xg,      // (NSEQ, LMAX, 1024)
              const __bf16* __restrict__ whh,    // (1024, 256) bf16 row-major
              const int* __restrict__ lens,      // per-sequence valid length
              float* __restrict__ out,           // (NSEQ, LMAX, 512)
              int backward, int outOff) {
  __shared__ __bf16 Habf[16][264];               // h state, bf16, padded
  __shared__ int larr[16];
  __shared__ int tarr[16];
  const int tid  = threadIdx.x;
  const int lane = tid & 31, wave = tid >> 5;
  const int hi = lane >> 4, cn = lane & 15;
  const int sb = blockIdx.x * 16;

  if (tid < 16) larr[tid] = lens[sb + tid];
  for (int i = tid; i < 16 * 264; i += 256) (&Habf[0][0])[i] = (__bf16)0.0f;
  int maxlen = 0;
  for (int i = 0; i < 16; ++i) {
    int l = lens[sb + i];
    maxlen = l > maxlen ? l : maxlen;
  }
  __syncthreads();

  f32x8 cst[2];
#pragma unroll
  for (int j2 = 0; j2 < 2; ++j2)
#pragma unroll
    for (int v = 0; v < 8; ++v) cst[j2][v] = 0.0f;
  const int jtab[2] = {wave, wave + 8};

  for (int s = 0; s < maxlen; ++s) {
    if (tid < 16) {
      int l = larr[tid];
      tarr[tid] = backward ? (l - 1 - s) : s;
    }
    __syncthreads();

    // init accumulators with gathered xg (garbage-free: masked by s<len)
    f32x8 acc[2][4];
#pragma unroll
    for (int j2 = 0; j2 < 2; ++j2) {
      const int jt = jtab[j2];
#pragma unroll
      for (int g = 0; g < 4; ++g) {
        const int ncol = g * 256 + jt * 16 + cn;
#pragma unroll
        for (int v = 0; v < 8; ++v) {
          const int m = v + 8 * hi;
          float x = 0.0f;
          if (s < larr[m])
            x = xg[(((size_t)(sb + m)) * LMAX + tarr[m]) * G4H + ncol];
          acc[j2][g][v] = x;
        }
      }
    }
    // G += H @ Whh^T   (K=256 in 8 steps of 32; 64 WMMAs per wave per step)
#pragma unroll
    for (int ks = 0; ks < 8; ++ks) {
      const int k0 = ks * 32;
      bf16x16 a = load_a_frag_bf16(&Habf[cn][k0], hi);
#pragma unroll
      for (int j2 = 0; j2 < 2; ++j2) {
        const int jt = jtab[j2];
#pragma unroll
        for (int g = 0; g < 4; ++g) {
          const int ncol = g * 256 + jt * 16 + cn;
          bf16x16 b = load_b_frag_bf16(whh + (size_t)ncol * HDIM + k0, hi);
          acc[j2][g] = wmma_bf16(a, b, acc[j2][g]);
        }
      }
    }
    __syncthreads();   // all Habf reads for this step done

    // cell update: c = sig(f)*c + sig(i)*tanh(g); h = sig(o)*tanh(c)
#pragma unroll
    for (int j2 = 0; j2 < 2; ++j2) {
      const int jt = jtab[j2];
      const int j = jt * 16 + cn;
#pragma unroll
      for (int v = 0; v < 8; ++v) {
        const int m = v + 8 * hi;
        const float iv = sigmf(acc[j2][0][v]);
        const float fv = sigmf(acc[j2][1][v]);
        const float gv = tanhf(acc[j2][2][v]);
        const float ov = sigmf(acc[j2][3][v]);
        const float c = fv * cst[j2][v] + iv * gv;
        cst[j2][v] = c;
        const float h = ov * tanhf(c);
        Habf[m][j] = (__bf16)h;
        if (s < larr[m])
          out[(((size_t)(sb + m)) * LMAX + tarr[m]) * DMODEL + outOff + j] = h;
      }
    }
    __syncthreads();
  }
}

// ---------------------------------------------------------------------------
// Fused attention core for one (seq, head, 32 q-rows):
//   S = (Q Kt)/sqrt(dh); S = mask ? S : -1e9; P = softmax(S); O = P V
// 128 threads (4 waves).  K buffer is reused (transposed) for V.
// ---------------------------------------------------------------------------
__global__ __launch_bounds__(128)
void attn_fused(const float* __restrict__ Qb, const float* __restrict__ Kb,
                const float* __restrict__ Vb, const int* __restrict__ kvmask,
                float* __restrict__ Cat) {
  __shared__ __bf16 Qlds[32][130];
  __shared__ __bf16 KVlds[128][130];
  __shared__ float  Slds[32][129];
  __shared__ int    msk[128];
  const int tid  = threadIdx.x;
  const int lane = tid & 31, wave = tid >> 5;
  const int hi = lane >> 4, cn = lane & 15;
  const int q0 = blockIdx.x * 32;
  const int head = blockIdx.y;
  const int n = blockIdx.z;
  const float* Qh = Qb + ((size_t)n * LMAX) * DMODEL + head * DHEAD;
  const float* Kh = Kb + ((size_t)n * LMAX) * DMODEL + head * DHEAD;
  const float* Vh = Vb + ((size_t)n * LMAX) * DMODEL + head * DHEAD;

  for (int i = tid; i < 32 * 128; i += 128) {
    const int r = i >> 7, c = i & 127;
    Qlds[r][c] = (__bf16)Qh[(size_t)(q0 + r) * DMODEL + c];
  }
  for (int i = tid; i < 128 * 128; i += 128) {
    const int r = i >> 7, c = i & 127;
    KVlds[r][c] = (__bf16)Kh[(size_t)r * DMODEL + c];
  }
  if (tid < 128) msk[tid] = kvmask[(size_t)n * LMAX + tid];
  __syncthreads();

  const int ms = wave & 1;
  const int nsb = (wave >> 1) * 4;

  // ---- S = Q @ K^T ----
  f32x8 acc[4];
#pragma unroll
  for (int q = 0; q < 4; ++q)
#pragma unroll
    for (int v = 0; v < 8; ++v) acc[q][v] = 0.0f;
#pragma unroll
  for (int ks = 0; ks < 4; ++ks) {
    const int k0 = ks * 32;
    bf16x16 a = load_a_frag_bf16(&Qlds[ms * 16 + cn][k0], hi);
#pragma unroll
    for (int q = 0; q < 4; ++q) {
      bf16x16 b = load_b_frag_bf16(&KVlds[(nsb + q) * 16 + cn][k0], hi);
      acc[q] = wmma_bf16(a, b, acc[q]);
    }
  }
  const float scale = 0.08838834764831843f;  // 1/sqrt(128)
#pragma unroll
  for (int q = 0; q < 4; ++q) {
    const int tcol = (nsb + q) * 16 + cn;
    const int mv = msk[tcol];
#pragma unroll
    for (int v = 0; v < 8; ++v) {
      const int mrow = ms * 16 + v + 8 * hi;
      Slds[mrow][tcol] = (mv != 0) ? acc[q][v] * scale : -1.0e9f;
    }
  }
  __syncthreads();

  // ---- load V transposed (reuse KVlds) + exact softmax on S ----
  for (int i = tid; i < 128 * 128; i += 128) {
    const int t = i >> 7, d = i & 127;
    KVlds[d][t] = (__bf16)Vh[(size_t)t * DMODEL + d];
  }
  if (tid < 32) {
    float mx = -3.0e38f;
    for (int t = 0; t < 128; ++t) mx = fmaxf(mx, Slds[tid][t]);
    float sum = 0.0f;
    for (int t = 0; t < 128; ++t) {
      const float e = __expf(Slds[tid][t] - mx);
      Slds[tid][t] = e;
      sum += e;
    }
    const float inv = 1.0f / sum;
    for (int t = 0; t < 128; ++t) Slds[tid][t] *= inv;
  }
  __syncthreads();

  // ---- O = P @ V ----
  f32x8 oacc[4];
#pragma unroll
  for (int q = 0; q < 4; ++q)
#pragma unroll
    for (int v = 0; v < 8; ++v) oacc[q][v] = 0.0f;
#pragma unroll
  for (int ks = 0; ks < 4; ++ks) {
    const int k0 = ks * 32;
    bf16x16 a = load_a_frag_f32(&Slds[ms * 16 + cn][k0], hi);
#pragma unroll
    for (int q = 0; q < 4; ++q) {
      bf16x16 b = load_b_frag_bf16(&KVlds[(nsb + q) * 16 + cn][k0], hi);
      oacc[q] = wmma_bf16(a, b, oacc[q]);
    }
  }
#pragma unroll
  for (int q = 0; q < 4; ++q) {
    const int dcol = (nsb + q) * 16 + cn;
#pragma unroll
    for (int v = 0; v < 8; ++v) {
      const int mrow = q0 + ms * 16 + v + 8 * hi;
      Cat[((size_t)n * LMAX + mrow) * DMODEL + head * DHEAD + dcol] = oacc[q][v];
    }
  }
}

// ---------------------------------------------------------------------------
// Small utility kernels
// ---------------------------------------------------------------------------
__global__ void zero_f32(float* __restrict__ p, long long count) {
  long long i = (long long)blockIdx.x * blockDim.x + threadIdx.x;
  const long long stride = (long long)gridDim.x * blockDim.x;
  for (; i < count; i += stride) p[i] = 0.0f;
}

__global__ void cvt_bf16(const float* __restrict__ s, __bf16* __restrict__ d,
                         int count) {
  int i = blockIdx.x * blockDim.x + threadIdx.x;
  const int stride = gridDim.x * blockDim.x;
  for (; i < count; i += stride) d[i] = (__bf16)s[i];
}

// ragged index + per-sequence masks/lengths
__global__ void prep_meta(const int* __restrict__ audio_mask,
                          const int* __restrict__ text_mask,
                          const int* __restrict__ audio_seqlens,
                          const int* __restrict__ text_seqlens,
                          int* __restrict__ ab, int* __restrict__ apos,
                          int* __restrict__ tb, int* __restrict__ tpos,
                          int* __restrict__ fmask, int* __restrict__ wmask,
                          int* __restrict__ flen, int* __restrict__ wlen) {
  const int n = blockIdx.x * blockDim.x + threadIdx.x;
  if (n >= NSEQ) return;
  {  // audio
    int cum = 0, b = 0, base = 0;
    for (int i = 0; i < BB; ++i) {
      const int sl = audio_seqlens[i];
      if (n < cum + sl) { b = i; base = cum; break; }
      cum += sl;
    }
    const int pos = n - base;
    ab[n] = b; apos[n] = pos;
    int cnt = 0;
    for (int t = 0; t < LMAX; ++t) {
      const int mv = audio_mask[((size_t)b * EE + pos) * LMAX + t];
      fmask[n * LMAX + t] = mv;
      cnt += (mv != 0);
    }
    flen[n] = cnt;
  }
  {  // text
    int cum = 0, b = 0, base = 0;
    for (int i = 0; i < BB; ++i) {
      const int sl = text_seqlens[i];
      if (n < cum + sl) { b = i; base = cum; break; }
      cum += sl;
    }
    const int pos = n - base;
    tb[n] = b; tpos[n] = pos;
    int cnt = 0;
    for (int t = 0; t < LMAX; ++t) {
      const int mv = text_mask[((size_t)b * EE + pos) * LMAX + t];
      wmask[n * LMAX + t] = mv;
      cnt += (mv != 0);
    }
    wlen[n] = cnt;
  }
}

// masked mean-pool (or BERT-style first token) + ragged scatter to (B,E,D)
__global__ void pool_pad(const float* __restrict__ X,
                         const int* __restrict__ lensArr,
                         const int* __restrict__ bi, const int* __restrict__ pi,
                         const int* __restrict__ bertFlag,
                         float* __restrict__ Out) {
  const int n = blockIdx.x;
  const int tid = threadIdx.x;
  const int useBert = bertFlag ? bertFlag[0] : 0;
  const int len = lensArr[n];
  const float* xb = X + (size_t)n * LMAX * DMODEL;
  const size_t obase = ((size_t)bi[n] * EE + pi[n]) * DMODEL;
  for (int d = tid; d < DMODEL; d += blockDim.x) {
    float v;
    if (useBert) {
      v = xb[d];  // x[:, 0]
    } else {
      float s = 0.0f;
      for (int t = 0; t < len; ++t) s += xb[(size_t)t * DMODEL + d];
      v = s / (float)len;
    }
    Out[obase + d] = v;
  }
}

// ---------------------------------------------------------------------------
// Host driver
// ---------------------------------------------------------------------------
extern "C" void kernel_launch(void* const* d_in, const int* in_sizes, int n_in,
                              void* d_out, int out_size, void* d_ws,
                              size_t ws_size, hipStream_t stream) {
  (void)n_in; (void)out_size; (void)ws_size;

  const float* x_text    = (const float*)d_in[0];
  const float* x_audio   = (const float*)d_in[1];
  const float* x_speaker = (const float*)d_in[2];

  // Nested-dict flattening detection (insertion order vs sorted keys).
  int a0 = 3, t0 = 9, f0 = 15, w0 = 19;     // group bases
  int oWihF = 0, oWhhF = 1, obF = 2, oWihB = 3, oWhhB = 4, obB = 5;  // insertion
  int oWq = 0, oWk = 1, oWv = 2, oWo = 3;
  if (in_sizes[3] == (G4H * HDIM)) {  // sorted: Whh_b,Whh_f,Wih_b,Wih_f,b_b,b_f
    oWhhB = 0; oWhhF = 1; oWihB = 2; oWihF = 3; obB = 4; obF = 5;
    oWk = 0; oWo = 1; oWq = 2; oWv = 3;        // sorted: Wk,Wo,Wq,Wv
    a0 = 3; f0 = 9; t0 = 13; w0 = 19;          // sorted params: audio,frame,text,word
  }
  const float* aWihF = (const float*)d_in[a0 + oWihF];
  const float* aWhhF = (const float*)d_in[a0 + oWhhF];
  const float* abF   = (const float*)d_in[a0 + obF];
  const float* aWihB = (const float*)d_in[a0 + oWihB];
  const float* aWhhB = (const float*)d_in[a0 + oWhhB];
  const float* abB   = (const float*)d_in[a0 + obB];
  const float* tWihF = (const float*)d_in[t0 + oWihF];
  const float* tWhhF = (const float*)d_in[t0 + oWhhF];
  const float* tbF   = (const float*)d_in[t0 + obF];
  const float* tWihB = (const float*)d_in[t0 + oWihB];
  const float* tWhhB = (const float*)d_in[t0 + oWhhB];
  const float* tbB   = (const float*)d_in[t0 + obB];
  const float* fWq = (const float*)d_in[f0 + oWq];
  const float* fWk = (const float*)d_in[f0 + oWk];
  const float* fWv = (const float*)d_in[f0 + oWv];
  const float* fWo = (const float*)d_in[f0 + oWo];
  const float* wWq = (const float*)d_in[w0 + oWq];
  const float* wWk = (const float*)d_in[w0 + oWk];
  const float* wWv = (const float*)d_in[w0 + oWv];
  const float* wWo = (const float*)d_in[w0 + oWo];

  const int* text_mask     = (const int*)d_in[23];
  const int* text_seqlens  = (const int*)d_in[25];
  const int* audio_mask    = (const int*)d_in[26];
  const int* audio_seqlens = (const int*)d_in[28];
  const int* use_bert      = (const int*)d_in[29];

  // ---- workspace layout ----
  char* W = (char*)d_ws;
  int* ab    = (int*)(W + 0);
  int* apos  = (int*)(W + 1024);
  int* tb    = (int*)(W + 2048);
  int* tpos  = (int*)(W + 3072);
  int* flen  = (int*)(W + 4096);
  int* wlen  = (int*)(W + 5120);
  int* fmask = (int*)(W + 8192);                    // NSEQ*LMAX ints
  int* wmask = (int*)(W + 8192 + 106496);
  __bf16* whhAF = (__bf16*)(W + 221184);            // 4 x (1024*256) bf16
  __bf16* whhAB = whhAF + (size_t)G4H * HDIM;
  __bf16* whhTF = whhAB + (size_t)G4H * HDIM;
  __bf16* whhTB = whhTF + (size_t)G4H * HDIM;

  const size_t SZ_XG = (size_t)MROWS * G4H * sizeof(float);     // 104 MB
  const size_t SZ_X  = (size_t)MROWS * DMODEL * sizeof(float);  // 52 MB
  const size_t BIGA = 4194304;
  float* xgF = (float*)(W + BIGA);
  float* xgB = (float*)(W + BIGA + SZ_XG);
  float* qb  = (float*)(W + BIGA);                  // reuses xg region later
  float* kb  = (float*)(W + BIGA + SZ_X);
  float* vb  = (float*)(W + BIGA + 2 * SZ_X);
  float* cat = (float*)(W + BIGA + 3 * SZ_X);
  const size_t BIGB = BIGA + 2 * SZ_XG;
  float* xa  = (float*)(W + BIGB);
  float* xs  = (float*)(W + BIGB + SZ_X);
  float* xt  = (float*)(W + BIGB + 2 * SZ_X);
  float* xtf = (float*)(W + BIGB + 3 * SZ_X);

  float* out = (float*)d_out;

  // ---- phase 0: meta, weight conversion, zero-fill ----
  prep_meta<<<1, 256, 0, stream>>>(audio_mask, text_mask, audio_seqlens,
                                   text_seqlens, ab, apos, tb, tpos, fmask,
                                   wmask, flen, wlen);
  cvt_bf16<<<512, 256, 0, stream>>>(aWhhF, whhAF, G4H * HDIM);
  cvt_bf16<<<512, 256, 0, stream>>>(aWhhB, whhAB, G4H * HDIM);
  cvt_bf16<<<512, 256, 0, stream>>>(tWhhF, whhTF, G4H * HDIM);
  cvt_bf16<<<512, 256, 0, stream>>>(tWhhB, whhTB, G4H * HDIM);
  zero_f32<<<4096, 256, 0, stream>>>(xa, (long long)4 * MROWS * DMODEL);
  zero_f32<<<256, 256, 0, stream>>>(out, (long long)4 * BB * EE * DMODEL);

  // ---- phase 1: three BiLSTMs (input-proj GEMMs + WMMA recurrence) ----
  const dim3 gP8(G4H / 128, MROWS / 128);   // N=1024 projections
  const dim3 gP4(DMODEL / 128, MROWS / 128);// N=512 projections
  struct Mod {
    const float* x; const float *WihF, *bF, *WihB, *bB;
    const __bf16 *whF, *whB; const int* lens; float* o;
  } mods[3] = {
      {x_audio,   aWihF, abF, aWihB, abB, whhAF, whhAB, flen, xa},
      {x_speaker, aWihF, abF, aWihB, abB, whhAF, whhAB, flen, xs},
      {x_text,    tWihF, tbF, tWihB, tbB, whhTF, whhTB, wlen, xt},
  };
  for (int mI = 0; mI < 3; ++mI) {
    const Mod& md = mods[mI];
    gemm_wmma<<<gP8, 256, 0, stream>>>(md.x, DMODEL, md.WihF, DMODEL, 1, md.bF,
                                       nullptr, nullptr, xgF, G4H, MROWS, G4H,
                                       DMODEL);
    gemm_wmma<<<gP8, 256, 0, stream>>>(md.x, DMODEL, md.WihB, DMODEL, 1, md.bB,
                                       nullptr, nullptr, xgB, G4H, MROWS, G4H,
                                       DMODEL);
    lstm_rec<<<NSEQ / 16, 256, 0, stream>>>(xgF, md.whF, md.lens, md.o, 0, 0);
    lstm_rec<<<NSEQ / 16, 256, 0, stream>>>(xgB, md.whB, md.lens, md.o, 1, HDIM);
  }

  // ---- phase 2: MHA1  xs += frame_mask ⊙ MHA(q=xs, kv=xt) ----
  const dim3 gAtt(LMAX / 32, NHEAD, NSEQ);
  gemm_wmma<<<gP4, 256, 0, stream>>>(xs, DMODEL, fWq, DMODEL, 0, nullptr,
                                     nullptr, nullptr, qb, DMODEL, MROWS,
                                     DMODEL, DMODEL);
  gemm_wmma<<<gP4, 256, 0, stream>>>(xt, DMODEL, fWk, DMODEL, 0, nullptr,
                                     nullptr, nullptr, kb, DMODEL, MROWS,
                                     DMODEL, DMODEL);
  gemm_wmma<<<gP4, 256, 0, stream>>>(xt, DMODEL, fWv, DMODEL, 0, nullptr,
                                     nullptr, nullptr, vb, DMODEL, MROWS,
                                     DMODEL, DMODEL);
  attn_fused<<<gAtt, 128, 0, stream>>>(qb, kb, vb, wmask, cat);
  gemm_wmma<<<gP4, 256, 0, stream>>>(cat, DMODEL, fWo, DMODEL, 0, nullptr,
                                     xs, fmask, xs, DMODEL, MROWS, DMODEL,
                                     DMODEL);

  // ---- phase 3: MHA2  xt_fuse = xt + word_mask ⊙ MHA(q=xt, kv=xs_new) ----
  gemm_wmma<<<gP4, 256, 0, stream>>>(xt, DMODEL, wWq, DMODEL, 0, nullptr,
                                     nullptr, nullptr, qb, DMODEL, MROWS,
                                     DMODEL, DMODEL);
  gemm_wmma<<<gP4, 256, 0, stream>>>(xs, DMODEL, wWk, DMODEL, 0, nullptr,
                                     nullptr, nullptr, kb, DMODEL, MROWS,
                                     DMODEL, DMODEL);
  gemm_wmma<<<gP4, 256, 0, stream>>>(xs, DMODEL, wWv, DMODEL, 0, nullptr,
                                     nullptr, nullptr, vb, DMODEL, MROWS,
                                     DMODEL, DMODEL);
  attn_fused<<<gAtt, 128, 0, stream>>>(qb, kb, vb, fmask, cat);
  gemm_wmma<<<gP4, 256, 0, stream>>>(cat, DMODEL, wWo, DMODEL, 0, nullptr,
                                     xt, wmask, xtf, DMODEL, MROWS, DMODEL,
                                     DMODEL);

  // ---- phase 4: pool + ragged scatter (output order: text, text_fuse,
  //               audio, speaker) ----
  const int stride = BB * EE * DMODEL;
  pool_pad<<<NSEQ, 256, 0, stream>>>(xt,  wlen, tb, tpos, use_bert, out);
  pool_pad<<<NSEQ, 256, 0, stream>>>(xtf, wlen, tb, tpos, use_bert,
                                     out + stride);
  pool_pad<<<NSEQ, 256, 0, stream>>>(xa,  flen, ab, apos, nullptr,
                                     out + 2 * stride);
  pool_pad<<<NSEQ, 256, 0, stream>>>(xs,  flen, ab, apos, nullptr,
                                     out + 3 * stride);
}
